// update_v_v1_1245540515926
// MI455X (gfx1250) — matrix operand order repaired and checked by reading, not compiled, not run on hardware
//
#include <hip/hip_runtime.h>

// ---- CDNA5 WMMA operand types ----
typedef __attribute__((ext_vector_type(16))) __bf16     v16bf;
typedef __attribute__((ext_vector_type(8)))  float      v8f;
typedef __attribute__((ext_vector_type(4)))  float      v4f;
typedef __attribute__((ext_vector_type(4)))  unsigned   v4u;

#define HID     128
#define EMB     256
#define NLAYERS 3

static __device__ __forceinline__ unsigned short f2bf(float f) {
    unsigned int u = __float_as_uint(f);
    unsigned int r = u + 0x7FFFu + ((u >> 16) & 1u);   // round-to-nearest-even
    return (unsigned short)(r >> 16);
}
static __device__ __forceinline__ float bf2f(unsigned short h) {
    return __uint_as_float(((unsigned int)h) << 16);
}

// A-fragment: built from two 16B LDS loads, bit-cast to v16bf (trivial types -> legal union)
union AFrag { v4u q[2]; v16bf v; };

// ---------------------------------------------------------------------------
// Scatter: vsum[i[e], :] += e2[e, :]   (one thread = one edge x 4 features)
// ---------------------------------------------------------------------------
__global__ __launch_bounds__(256) void scatter_kernel(
    const float* __restrict__ e2, const int* __restrict__ idx,
    float* __restrict__ vsum, int E)
{
    long long t = (long long)blockIdx.x * blockDim.x + threadIdx.x;
    int edge = (int)(t >> 5);
    int f4   = (int)(t & 31);
    if (edge >= E) return;
    v4f val = reinterpret_cast<const v4f*>(e2 + (size_t)edge * HID)[f4];
    int node = idx[edge];
    float* dst = vsum + (size_t)node * HID + f4 * 4;
    atomicAdd(dst + 0, val[0]);
    atomicAdd(dst + 1, val[1]);
    atomicAdd(dst + 2, val[2]);
    atomicAdd(dst + 3, val[3]);
}

// ---------------------------------------------------------------------------
// Swizzle weights (f32 row-major [K, EMB]) into bf16 WMMA B-fragment order:
//   dst[((kt*16 + nt)*32 + lane)*16 + e] = W[(kt*32 + (e&7) + 8*(lane>=16) + 16*(e>=8))*EMB + nt*16 + (lane&15)]
// (ISA 7.12.2: 16-bit B 32x16 — e<8 -> K=8*h+e, e>=8 -> K=16+8*h+(e-8), matching A)
// ---------------------------------------------------------------------------
__global__ __launch_bounds__(256) void swizzle_kernel(
    const float* __restrict__ W, unsigned short* __restrict__ dst,
    int ktiles, int elems_per_layer)
{
    int t = blockIdx.x * 256 + threadIdx.x;
    int layer = t / elems_per_layer;
    int r = t - layer * elems_per_layer;
    int e  = r & 15;
    int L  = (r >> 4) & 31;
    int nt = (r >> 9) & 15;
    int kt = r >> 13;
    int K  = kt * 32 + (e & 7) + ((L >> 4) << 3) + ((e >> 3) << 4);
    int N  = nt * 16 + (L & 15);
    dst[t] = f2bf(W[(size_t)layer * (size_t)(ktiles * 32) * EMB + (size_t)K * EMB + N]);
}

// ---------------------------------------------------------------------------
// Fused MLP: one workgroup (8 wave32) per 16-node tile.
//   up:   [16,128] @ [128,256] + b_up
//   3x:   silu([16,256] @ [256,256] + b)
//   out:  [16,256] @ [256,1]
// ---------------------------------------------------------------------------
__global__ __launch_bounds__(256) void mlp_kernel(
    const float* __restrict__ vsum,
    const float* __restrict__ b_up,
    const float* __restrict__ b_lins,          // [3][256]
    const float* __restrict__ W_out,           // [256]
    const unsigned short* __restrict__ wup_sw, // bf16, fragment order
    const unsigned short* __restrict__ wlin_sw,
    float* __restrict__ out, int N)
{
    __shared__ alignas(16) unsigned short bufA[16][EMB];
    __shared__ alignas(16) unsigned short bufB[16][EMB];

    const int tid  = threadIdx.x;
    const int wave = tid >> 5;
    const int lane = tid & 31;
    const int half = lane >> 4;       // lane group (ISA K/M split)
    const int mcol = lane & 15;
    const int node0 = blockIdx.x * 16;
    const int nt0 = 2 * wave;         // each wave owns 2 N-tiles (32 cols)

    // cooperative load vsum tile [16,128] f32 -> bf16 into bufA[:,0:128]
    {
        int row  = tid >> 4;
        int col0 = (tid & 15) * 8;
        v4f a = {0.f, 0.f, 0.f, 0.f}, b = {0.f, 0.f, 0.f, 0.f};
        if (node0 + row < N) {
            const v4f* p = reinterpret_cast<const v4f*>(vsum + (size_t)(node0 + row) * HID + col0);
            a = p[0]; b = p[1];
        }
        unsigned short* d = &bufA[row][col0];
        d[0] = f2bf(a[0]); d[1] = f2bf(a[1]); d[2] = f2bf(a[2]); d[3] = f2bf(a[3]);
        d[4] = f2bf(b[0]); d[5] = f2bf(b[1]); d[6] = f2bf(b[2]); d[7] = f2bf(b[3]);
    }
    __syncthreads();

    v8f acc0 = {0.f,0.f,0.f,0.f,0.f,0.f,0.f,0.f};
    v8f acc1 = acc0;

    // ---- lin_up: K = 128 (4 k-steps of 32) ----
#pragma unroll
    for (int kt = 0; kt < HID / 32; ++kt) {
        AFrag af;
        int c0 = kt * 32 + 8 * half;   // ISA 16-bit A: e<8 -> K=c0+e ; e>=8 -> K=c0+16+(e-8)
        af.q[0] = *reinterpret_cast<const v4u*>(&bufA[mcol][c0]);
        af.q[1] = *reinterpret_cast<const v4u*>(&bufA[mcol][c0 + 16]);
        v16bf b0 = *reinterpret_cast<const v16bf*>(wup_sw + (((size_t)(kt * 16 + nt0    ) * 32 + lane) << 4));
        v16bf b1 = *reinterpret_cast<const v16bf*>(wup_sw + (((size_t)(kt * 16 + nt0 + 1) * 32 + lane) << 4));
        acc0 = __builtin_amdgcn_wmma_f32_16x16x32_bf16(false, af.v, false, b0, (short)0, acc0, false, false);
        acc1 = __builtin_amdgcn_wmma_f32_16x16x32_bf16(false, af.v, false, b1, (short)0, acc1, false, false);
    }
    {
        float bias0 = b_up[nt0 * 16 + mcol];
        float bias1 = b_up[nt0 * 16 + 16 + mcol];
#pragma unroll
        for (int r = 0; r < 8; ++r) {
            int rowM = r + 8 * half;   // ISA f32 C/D: VGPR r -> M = r + 8*(lane>=16)
            bufB[rowM][nt0 * 16 + mcol]      = f2bf(acc0[r] + bias0);
            bufB[rowM][nt0 * 16 + 16 + mcol] = f2bf(acc1[r] + bias1);
        }
    }
    __syncthreads();

    unsigned short (*cur)[EMB] = bufB;
    unsigned short (*nxt)[EMB] = bufA;

    // ---- 3 hidden layers: K = 256 (8 k-steps), bias + SiLU ----
    for (int l = 0; l < NLAYERS; ++l) {
        acc0 = (v8f){0.f,0.f,0.f,0.f,0.f,0.f,0.f,0.f};
        acc1 = acc0;
        const unsigned short* wl = wlin_sw + (size_t)l * (EMB / 32) * 16 * 32 * 16;
#pragma unroll
        for (int kt = 0; kt < EMB / 32; ++kt) {
            AFrag af;
            int c0 = kt * 32 + 8 * half;
            af.q[0] = *reinterpret_cast<const v4u*>(&cur[mcol][c0]);
            af.q[1] = *reinterpret_cast<const v4u*>(&cur[mcol][c0 + 16]);
            v16bf b0 = *reinterpret_cast<const v16bf*>(wl + (((size_t)(kt * 16 + nt0    ) * 32 + lane) << 4));
            v16bf b1 = *reinterpret_cast<const v16bf*>(wl + (((size_t)(kt * 16 + nt0 + 1) * 32 + lane) << 4));
            acc0 = __builtin_amdgcn_wmma_f32_16x16x32_bf16(false, af.v, false, b0, (short)0, acc0, false, false);
            acc1 = __builtin_amdgcn_wmma_f32_16x16x32_bf16(false, af.v, false, b1, (short)0, acc1, false, false);
        }
        float bias0 = b_lins[l * EMB + nt0 * 16 + mcol];
        float bias1 = b_lins[l * EMB + nt0 * 16 + 16 + mcol];
#pragma unroll
        for (int r = 0; r < 8; ++r) {
            int rowM = r + 8 * half;
            float x0 = acc0[r] + bias0;
            float x1 = acc1[r] + bias1;
            nxt[rowM][nt0 * 16 + mcol]      = f2bf(x0 / (1.0f + __expf(-x0)));
            nxt[rowM][nt0 * 16 + 16 + mcol] = f2bf(x1 / (1.0f + __expf(-x1)));
        }
        unsigned short (*tmp)[EMB] = cur; cur = nxt; nxt = tmp;
        __syncthreads();
    }

    // ---- final [16,256] @ [256,1]: f32 dot + 16-lane shuffle reduce ----
    {
        int m   = tid >> 4;
        int seg = tid & 15;
        float p = 0.0f;
#pragma unroll
        for (int j = 0; j < 16; ++j) {
            int k = seg * 16 + j;
            p += bf2f(cur[m][k]) * W_out[k];
        }
#pragma unroll
        for (int off = 8; off > 0; off >>= 1)
            p += __shfl_xor(p, off, 32);   // stays within each 16-lane node group
        if (seg == 0 && node0 + m < N) out[node0 + m] = p;
    }
}

// ---------------------------------------------------------------------------
extern "C" void kernel_launch(void* const* d_in, const int* in_sizes, int n_in,
                              void* d_out, int out_size, void* d_ws, size_t ws_size,
                              hipStream_t stream)
{
    // inputs: e1, e2, i, num_nodes, W_up, b_up, W_lins, b_lins, W_out
    const float* e2     = (const float*)d_in[1];
    const int*   idx    = (const int*)d_in[2];
    const float* W_up   = (const float*)d_in[4];
    const float* b_up   = (const float*)d_in[5];
    const float* W_lins = (const float*)d_in[6];
    const float* b_lins = (const float*)d_in[7];
    const float* W_out  = (const float*)d_in[8];
    float* out = (float*)d_out;

    const int N = out_size;             // OUT == 1 -> out_size == num_nodes
    const int E = in_sizes[1] / HID;

    // workspace layout
    char* ws = (char*)d_ws;
    unsigned short* wup_sw  = (unsigned short*)ws;                 // 32768 bf16   (64 KB)
    unsigned short* wlin_sw = (unsigned short*)(ws + 65536);       // 3*65536 bf16 (384 KB)
    float* vsum             = (float*)(ws + 65536 + 393216);       // N*128 f32

    hipMemsetAsync(vsum, 0, (size_t)N * HID * sizeof(float), stream);

    const int up_elems  = (HID / 32) * 16 * 32 * 16;   // 32768
    const int lin_elems = (EMB / 32) * 16 * 32 * 16;   // 65536
    swizzle_kernel<<<up_elems / 256, 256, 0, stream>>>(W_up, wup_sw, HID / 32, up_elems);
    swizzle_kernel<<<(NLAYERS * lin_elems) / 256, 256, 0, stream>>>(W_lins, wlin_sw, EMB / 32, lin_elems);

    long long sthreads = (long long)E * 32;
    scatter_kernel<<<(unsigned)((sthreads + 255) / 256), 256, 0, stream>>>(e2, idx, vsum, E);

    mlp_kernel<<<(N + 15) / 16, 256, 0, stream>>>(vsum, b_up, b_lins, W_out, wup_sw, wlin_sw, out, N);
}
